// AttentionBlock_34119220199648
// MI455X (gfx1250) — matrix operand order repaired and checked by reading, not compiled, not run on hardware
//
#include <hip/hip_runtime.h>

typedef __bf16 bf16;
typedef __attribute__((ext_vector_type(16))) __bf16 v16bf;
typedef __attribute__((ext_vector_type(8)))  __bf16 bf16x8;
typedef __attribute__((ext_vector_type(2)))  __bf16 bf16x2;
typedef __attribute__((ext_vector_type(8)))  float  v8f;
typedef __attribute__((ext_vector_type(4)))  unsigned int u32x4;
typedef __attribute__((ext_vector_type(8)))  int i32x8;
typedef __attribute__((ext_vector_type(4)))  int i32x4;

#define CB 512
#define NB 1024
#define QKVC 1536
#define LDSP 40   // padded LDS row stride (elements): 80 B rows, 16-B aligned

// ---------------- TDM helpers (device pass only) ----------------
__device__ __forceinline__ void wait_tensorcnt0() {
#if defined(__HIP_DEVICE_COMPILE__)
    __builtin_amdgcn_s_wait_tensorcnt(0);
#endif
}

// 2D row-major tile (tile_k elems wide, tile_rows rows) -> LDS with 16B pad per 64B row.
// tile_k must be 32 (64-byte rows) to match pad_interval/pad_amount below.
__device__ __forceinline__ void tdm_load_2d(unsigned lds_byte_addr, const void* gaddr,
                                            unsigned tile_k, unsigned tile_rows,
                                            unsigned ld_elems, unsigned tensor_rows) {
#if defined(__HIP_DEVICE_COMPILE__)
    unsigned long long ga = (unsigned long long)gaddr;
    u32x4 g0;
    g0[0] = 1u;                                        // count=1 (valid), user mode
    g0[1] = lds_byte_addr;                             // lds_addr  (bits 63:32)
    g0[2] = (unsigned)(ga & 0xffffffffu);              // global_addr[31:0]
    g0[3] = (unsigned)((ga >> 32) & 0x1ffffffu)        // global_addr[56:32]
          | (2u << 30);                                // type=2 ("image")
    i32x8 g1;
    g1[0] = (int)((1u << 16)                           // data_size = 2 bytes
                | (1u << 20)                           // pad_enable
                | (3u << 22)                           // pad_interval: 64 bytes
                | (3u << 25));                         // pad_amount: 4 DWORDs (16 B)
    g1[1] = (int)((ld_elems & 0xffffu) << 16);         // tensor_dim0[15:0]
    g1[2] = (int)(((ld_elems >> 16) & 0xffffu)         // tensor_dim0[31:16]
                | ((tensor_rows & 0xffffu) << 16));    // tensor_dim1[15:0]
    g1[3] = (int)(((tensor_rows >> 16) & 0xffffu)      // tensor_dim1[31:16]
                | (tile_k << 16));                     // tile_dim0
    g1[4] = (int)(tile_rows & 0xffffu);                // tile_dim1 (tile_dim2 = 0)
    g1[5] = (int)ld_elems;                             // tensor_dim0_stride[31:0]
    g1[6] = 0;                                         // stride hi / dim1_stride lo
    g1[7] = 0;
    i32x4 z4 = {0, 0, 0, 0};
#if __clang_major__ >= 23
    i32x8 z8 = {0, 0, 0, 0, 0, 0, 0, 0};
    __builtin_amdgcn_tensor_load_to_lds(g0, g1, z4, z4, z8, 0);
#else
    __builtin_amdgcn_tensor_load_to_lds(g0, g1, z4, z4, 0);
#endif
#else
    (void)lds_byte_addr; (void)gaddr; (void)tile_k; (void)tile_rows;
    (void)ld_elems; (void)tensor_rows;
#endif
}

__device__ __forceinline__ unsigned lds_addr_of(const void* p) {
    return (unsigned)(size_t)p;   // low 32 bits of generic pointer = LDS offset
}

// ---------------- GroupNorm ----------------
__global__ void gn_stats_kernel(const float* __restrict__ x, float* __restrict__ stats) {
    __shared__ float s_sum[256], s_sq[256];
    int blk = blockIdx.x;           // b*32 + g
    int b = blk >> 5, g = blk & 31;
    const float* base = x + (size_t)b * CB * NB + (size_t)g * 16 * NB;
    float sum = 0.f, sq = 0.f;
    for (int i = threadIdx.x; i < 16 * NB; i += 256) {
        float v = base[i]; sum += v; sq += v * v;
    }
    s_sum[threadIdx.x] = sum; s_sq[threadIdx.x] = sq;
    __syncthreads();
    for (int s = 128; s > 0; s >>= 1) {
        if (threadIdx.x < s) {
            s_sum[threadIdx.x] += s_sum[threadIdx.x + s];
            s_sq[threadIdx.x]  += s_sq[threadIdx.x + s];
        }
        __syncthreads();
    }
    if (threadIdx.x == 0) {
        float mean = s_sum[0] * (1.0f / (16 * NB));
        float var  = s_sq[0] * (1.0f / (16 * NB)) - mean * mean;
        stats[blk * 2]     = mean;
        stats[blk * 2 + 1] = rsqrtf(var + 1e-5f);
    }
}

__global__ void gn_apply_kernel(const float* __restrict__ x, const float* __restrict__ gamma,
                                const float* __restrict__ beta, const float* __restrict__ stats,
                                bf16* __restrict__ xn) {
    size_t idx = (size_t)blockIdx.x * 256 + threadIdx.x;
    size_t b = idx >> 19;                 // /(512*1024)
    int c = (int)((idx >> 10) & 511);
    int g = c >> 4;
    int sidx = ((int)b * 32 + g) * 2;
    float mean = stats[sidx], rstd = stats[sidx + 1];
    float v = (x[idx] - mean) * rstd * gamma[c] + beta[c];
    xn[idx] = (bf16)v;
}

__global__ void f32_to_bf16_kernel(const float* __restrict__ src, bf16* __restrict__ dst, int n) {
    int i = blockIdx.x * 256 + threadIdx.x;
    if (i < n) dst[i] = (bf16)src[i];
}

// ---------------- Generic WMMA bf16 GEMM ----------------
// MODE 0: qkv = w_in * xn          (+bias, scale q rows by 1/sqrt(512), bf16 out)
// MODE 1: S   = Q^T * K            (A source stored [K x M], f32 out)
// MODE 2: O   = V * attn^T         (B source stored [N x K], bf16 out)
// MODE 3: out = w_out * O + b + x  (f32 out with skip)
template <int MODE>
__global__ __launch_bounds__(256) void gemm_bf16_kernel(
    const bf16* __restrict__ Aall, const bf16* __restrict__ Ball, void* __restrict__ Dall,
    const float* __restrict__ bias, const float* __restrict__ skip,
    int M, int N, int K, int ldA, int ldB,
    long aStride, long bStride, long dStride, long skipStride)
{
    __shared__ bf16 ldsA[128 * LDSP];   // [m][k], 80-byte padded rows
    __shared__ bf16 ldsB[64 * LDSP];    // [n][k]

    const int tid   = threadIdx.x;
    const int wave  = tid >> 5;
    const int lane  = tid & 31;
    const int l16   = lane & 15;
    const int half  = lane >> 4;
    const int batch = blockIdx.z;
    const int nbase = blockIdx.x * 64;
    const int mbase = blockIdx.y * 128;

    const bf16* A = Aall + (size_t)batch * aStride;
    const bf16* B = Ball + (size_t)batch * bStride;

    v8f acc[4];
#pragma unroll
    for (int i = 0; i < 4; ++i)
#pragma unroll
        for (int r = 0; r < 8; ++r) acc[i][r] = 0.0f;

    for (int kb = 0; kb < K; kb += 32) {
        // ---- stage A tile (128 x 32) ----
        if (MODE == 1) {   // A source [K x M]: transpose while writing to LDS
#pragma unroll
            for (int i = 0; i < 8; ++i) {
                int lin = tid + i * 256;
                int r = lin >> 6, c2 = lin & 63;
                bf16x2 p = *(const bf16x2*)(A + (size_t)(kb + r) * ldA + mbase + c2 * 2);
                ldsA[(c2 * 2 + 0) * LDSP + r] = p.x;
                ldsA[(c2 * 2 + 1) * LDSP + r] = p.y;
            }
        } else {           // A source [M x K] row-major: TDM DMA with LDS row padding
            if (wave == 0)
                tdm_load_2d(lds_addr_of(ldsA),
                            A + (size_t)mbase * ldA + kb,
                            32u, 128u, (unsigned)ldA, (unsigned)M);
        }
        // ---- stage B tile into [n][k] layout ----
        if (MODE == 2) {   // B source [N x K] row-major: TDM DMA straight into [n][k]
            if (wave == 1)
                tdm_load_2d(lds_addr_of(ldsB),
                            B + (size_t)nbase * ldB + kb,
                            32u, 64u, (unsigned)ldB, (unsigned)N);
        } else {           // B source [K x N]: transpose into [n][k]
#pragma unroll
            for (int i = 0; i < 4; ++i) {
                int lin = tid + i * 256;
                int r = lin >> 5, c2 = lin & 31;
                bf16x2 p = *(const bf16x2*)(B + (size_t)(kb + r) * ldB + nbase + c2 * 2);
                ldsB[(c2 * 2 + 0) * LDSP + r] = p.x;
                ldsB[(c2 * 2 + 1) * LDSP + r] = p.y;
            }
        }
        if (MODE != 1 && wave == 0) wait_tensorcnt0();
        if (MODE == 2 && wave == 1) wait_tensorcnt0();
        __syncthreads();

        // A fragment: lane holds K = half*8 + {0..7} and 16 + half*8 + {0..7} (ISA 7.12.2)
        const bf16* arow = &ldsA[(wave * 16 + l16) * LDSP + half * 8];
        bf16x8 alo = *(const bf16x8*)(arow);
        bf16x8 ahi = *(const bf16x8*)(arow + 16);
        v16bf afrag = __builtin_shufflevector(alo, ahi,
            0, 1, 2, 3, 4, 5, 6, 7, 8, 9, 10, 11, 12, 13, 14, 15);
#pragma unroll
        for (int nf = 0; nf < 4; ++nf) {
            // B fragment: lane holds K = half*8 + {0..7} and 16 + half*8 + {0..7}, same as A
            const bf16* brow = &ldsB[(nf * 16 + l16) * LDSP + half * 8];
            bf16x8 blo = *(const bf16x8*)(brow);
            bf16x8 bhi = *(const bf16x8*)(brow + 16);
            v16bf bfrag = __builtin_shufflevector(blo, bhi,
                0, 1, 2, 3, 4, 5, 6, 7, 8, 9, 10, 11, 12, 13, 14, 15);
            acc[nf] = __builtin_amdgcn_wmma_f32_16x16x32_bf16(
                false, afrag, false, bfrag, (short)0, acc[nf], false, false);
        }
        __syncthreads();
    }

    // ---- epilogue (C/D f32 16x16 layout: lane = N, vgpr r + half*8 = M) ----
#pragma unroll
    for (int nf = 0; nf < 4; ++nf) {
        int n = nbase + nf * 16 + l16;
#pragma unroll
        for (int r = 0; r < 8; ++r) {
            int m = mbase + wave * 16 + half * 8 + r;
            float v = acc[nf][r];
            size_t di = (size_t)batch * dStride + (size_t)m * N + n;
            if (MODE == 0) {
                v += bias[m];
                if (m < 512) v *= 0.044194173824159216f;   // 1/sqrt(512) folded into q
                ((bf16*)Dall)[di] = (bf16)v;
            } else if (MODE == 1) {
                ((float*)Dall)[di] = v;
            } else if (MODE == 2) {
                ((bf16*)Dall)[di] = (bf16)v;
            } else {
                v += bias[m] + skip[(size_t)batch * skipStride + (size_t)m * N + n];
                ((float*)Dall)[di] = v;
            }
        }
    }
}

// ---------------- Softmax over 1024-wide rows ----------------
__global__ void softmax_kernel(const float* __restrict__ S, bf16* __restrict__ A) {
    __shared__ float red[256];
    size_t row = blockIdx.x;
    const float* src = S + row * NB;
    bf16* dst = A + row * NB;
    float vals[4];
    float mx = -1e30f;
#pragma unroll
    for (int i = 0; i < 4; ++i) { vals[i] = src[threadIdx.x + i * 256]; mx = fmaxf(mx, vals[i]); }
    red[threadIdx.x] = mx; __syncthreads();
    for (int s = 128; s > 0; s >>= 1) {
        if (threadIdx.x < s) red[threadIdx.x] = fmaxf(red[threadIdx.x], red[threadIdx.x + s]);
        __syncthreads();
    }
    mx = red[0]; __syncthreads();
    float sum = 0.f;
#pragma unroll
    for (int i = 0; i < 4; ++i) { vals[i] = __expf(vals[i] - mx); sum += vals[i]; }
    red[threadIdx.x] = sum; __syncthreads();
    for (int s = 128; s > 0; s >>= 1) {
        if (threadIdx.x < s) red[threadIdx.x] += red[threadIdx.x + s];
        __syncthreads();
    }
    float inv = 1.0f / red[0];
#pragma unroll
    for (int i = 0; i < 4; ++i) dst[threadIdx.x + i * 256] = (bf16)(vals[i] * inv);
}

// ---------------- host launcher ----------------
extern "C" void kernel_launch(void* const* d_in, const int* in_sizes, int n_in,
                              void* d_out, int out_size, void* d_ws, size_t ws_size,
                              hipStream_t stream) {
    const float* x     = (const float*)d_in[0];
    const float* gamma = (const float*)d_in[1];
    const float* beta  = (const float*)d_in[2];
    const float* w_in  = (const float*)d_in[3];
    const float* b_in  = (const float*)d_in[4];
    const float* w_out = (const float*)d_in[5];
    const float* b_out = (const float*)d_in[6];

    char* ws = (char*)d_ws;
    size_t off = 0;
    bf16*  w_in_bf  = (bf16*)(ws + off);  off += (size_t)QKVC * CB * 2;
    bf16*  w_out_bf = (bf16*)(ws + off);  off += (size_t)CB * CB * 2;
    float* stats    = (float*)(ws + off); off += (size_t)512 * 2 * 4;
    bf16*  xn       = (bf16*)(ws + off);  off += (size_t)16 * CB * NB * 2;
    bf16*  qkv      = (bf16*)(ws + off);  off += (size_t)16 * QKVC * NB * 2;
    float* S        = (float*)(ws + off); off += (size_t)16 * NB * NB * 4;
    bf16*  attn     = (bf16*)(ws + off);  off += (size_t)16 * NB * NB * 2;
    bf16*  O        = (bf16*)(ws + off);  off += (size_t)16 * CB * NB * 2;
    (void)ws_size; (void)in_sizes; (void)n_in; (void)out_size;

    f32_to_bf16_kernel<<<(QKVC * CB + 255) / 256, 256, 0, stream>>>(w_in, w_in_bf, QKVC * CB);
    f32_to_bf16_kernel<<<(CB * CB + 255) / 256, 256, 0, stream>>>(w_out, w_out_bf, CB * CB);
    gn_stats_kernel<<<512, 256, 0, stream>>>(x, stats);
    gn_apply_kernel<<<(16 * CB * NB) / 256, 256, 0, stream>>>(x, gamma, beta, stats, xn);

    // qkv[b] = w_in (1536x512) * xn[b] (512x1024)   [q rows pre-scaled]
    gemm_bf16_kernel<0><<<dim3(NB / 64, QKVC / 128, 16), 256, 0, stream>>>(
        w_in_bf, xn, qkv, b_in, nullptr,
        QKVC, NB, CB, CB, NB,
        0L, (long)CB * NB, (long)QKVC * NB, 0L);

    // S[b] = Q[b]^T * K[b]  (1024x1024), Q/K stored [512 x 1024] inside qkv
    gemm_bf16_kernel<1><<<dim3(NB / 64, NB / 128, 16), 256, 0, stream>>>(
        qkv, qkv + (size_t)CB * NB, S, nullptr, nullptr,
        NB, NB, CB, NB, NB,
        (long)QKVC * NB, (long)QKVC * NB, (long)NB * NB, 0L);

    softmax_kernel<<<16 * NB, 256, 0, stream>>>(S, attn);

    // O[b] = V[b] (512x1024) * attn[b]^T (1024x1024)
    gemm_bf16_kernel<2><<<dim3(NB / 64, CB / 128, 16), 256, 0, stream>>>(
        qkv + (size_t)2 * CB * NB, attn, O, nullptr, nullptr,
        CB, NB, NB, NB, NB,
        (long)QKVC * NB, (long)NB * NB, (long)CB * NB, 0L);

    // out[b] = w_out (512x512) * O[b] + b_out + x[b]
    gemm_bf16_kernel<3><<<dim3(NB / 64, CB / 128, 16), 256, 0, stream>>>(
        w_out_bf, O, d_out, b_out, x,
        CB, NB, CB, CB, NB,
        0L, (long)CB * NB, (long)CB * NB, (long)CB * NB);
}
